// SepnetLinear_200_31636729102796
// MI455X (gfx1250) — matrix operand order repaired and checked by reading, not compiled, not run on hardware
//
#include <hip/hip_runtime.h>
#include <hip/hip_bf16.h>
#include <stdint.h>

typedef __attribute__((ext_vector_type(16))) __bf16 v16bf;
typedef __attribute__((ext_vector_type(8)))  float  v8f;

union Frag16 { v16bf v; uint4 q[2]; };

// ---------------- async global->LDS availability ----------------
#if defined(__has_builtin)
#  if __has_builtin(__builtin_amdgcn_global_load_async_to_lds_b128) && \
      __has_builtin(__builtin_amdgcn_s_wait_asynccnt)
#    define USE_ASYNC_COPY 1
#  endif
#endif
#ifndef USE_ASYNC_COPY
#  define USE_ASYNC_COPY 0
#endif

#if USE_ASYNC_COPY
// builtin expects: (v4i addrspace(1)*, v4i addrspace(3)*, imm offset, imm cpol)
typedef int v4i __attribute__((vector_size(4 * sizeof(int))));
typedef __attribute__((address_space(1))) v4i as1_v4i;
typedef __attribute__((address_space(3))) v4i as3_v4i;
__device__ __forceinline__ as1_v4i* to_global(const void* p) {
  return (as1_v4i*)(uintptr_t)p;
}
__device__ __forceinline__ as3_v4i* to_lds(const void* p) {
  // flat shared pointer: addr[31:0] is the LDS byte address (ISA aperture rule)
  return (as3_v4i*)(uintptr_t)(uint32_t)(uintptr_t)p;
}
#endif

// ---------------- problem dims ----------------
#define DIM_B   32
#define DIM_C   256
#define DIM_H   4096
#define DIM_AW  1024
#define DIM_BH  512

#define NA (DIM_AW * DIM_H)          // 4194304
#define NB (DIM_C * DIM_BH)          // 131072
#define NX (DIM_B * DIM_C * DIM_H)   // 33554432
#define JA ((unsigned)(NA / 2))
#define JB ((unsigned)(NB / 2))

// ---------------- workspace layout (bytes) ----------------
#define WS_REFA      ((size_t)0)                                   // bf16 [1024][4096]
#define WS_XBF       (WS_REFA  + (size_t)NA * 2)                   // bf16 [32][256][4096]
#define WS_REFBT     (WS_XBF   + (size_t)NX * 2)                   // bf16 [512][256] (transposed)
#define WS_TEMP      (WS_REFBT + (size_t)NB * 2)                   // f32  [32*1024][256]
#define WS_HIST_AHI  (WS_TEMP  + (size_t)DIM_B * DIM_AW * DIM_C * 4)
#define WS_HIST_ALO  (WS_HIST_AHI + (size_t)65536 * 4)
#define WS_HIST_BHI  (WS_HIST_ALO + (size_t)65536 * 4)
#define WS_HIST_BLO  (WS_HIST_BHI + (size_t)65536 * 4)
#define WS_SCAL      (WS_HIST_BLO + (size_t)65536 * 4)
#define WS_ZERO_BYTES ((size_t)65536 * 4 * 4 + 64)

// ---------------- output layout (floats) ----------------
#define OUT_ACOPY ((size_t)DIM_B * DIM_AW * DIM_BH)   // 16777216
#define OUT_BCOPY (OUT_ACOPY + (size_t)NA)            // 20971520

// ============ radix-select (top-half of |scores|) ============

__global__ void hist_hi(const float* __restrict__ s, int n, unsigned* __restrict__ hist) {
  for (int i = blockIdx.x * blockDim.x + threadIdx.x; i < n; i += gridDim.x * blockDim.x) {
    unsigned bits = __float_as_uint(fabsf(s[i]));
    atomicAdd(&hist[bits >> 16], 1u);
  }
}

__global__ void hist_lo(const float* __restrict__ s, int n,
                        const unsigned* __restrict__ bin_base, unsigned* __restrict__ hist) {
  const unsigned bin = bin_base[0];
  for (int i = blockIdx.x * blockDim.x + threadIdx.x; i < n; i += gridDim.x * blockDim.x) {
    unsigned bits = __float_as_uint(fabsf(s[i]));
    if ((bits >> 16) == bin) atomicAdd(&hist[bits & 0xFFFFu], 1u);
  }
}

__global__ void select_scan(const unsigned* __restrict__ hist, unsigned j,
                            const unsigned* __restrict__ bin_base_in,
                            unsigned* __restrict__ result) {
  __shared__ unsigned part[64];
  const int t = threadIdx.x;  // 64 threads
  unsigned sum = 0;
  for (int i = 0; i < 1024; ++i) sum += hist[t * 1024 + i];
  part[t] = sum;
  __syncthreads();
  if (t == 0) {
    unsigned running = (bin_base_in != nullptr) ? bin_base_in[1] : 0u;
    for (int c = 0; c < 64; ++c) {
      if (running + part[c] > j) {
        unsigned r2 = running;
        unsigned bin = (unsigned)(c * 1024 + 1023);
        for (int i = 0; i < 1024; ++i) {
          unsigned cnt = hist[c * 1024 + i];
          if (r2 + cnt > j) { bin = (unsigned)(c * 1024 + i); break; }
          r2 += cnt;
        }
        if (bin_base_in == nullptr) { result[0] = bin; result[1] = r2; }
        else                        { result[0] = (bin_base_in[0] << 16) | bin; }
        return;
      }
      running += part[c];
    }
    if (bin_base_in == nullptr) { result[0] = 65535u; result[1] = running; }
    else                        { result[0] = (bin_base_in[0] << 16) | 65535u; }
  }
}

// ============ conversions / masking (+ passthrough copies) ============

__device__ __forceinline__ unsigned pack_bf16(float lo, float hi) {
  return ((unsigned)__builtin_bit_cast(unsigned short, (__bf16)hi) << 16) |
         (unsigned)__builtin_bit_cast(unsigned short, (__bf16)lo);
}

__global__ void convert_x_bf16(const float* __restrict__ x, __bf16* __restrict__ xbf) {
  size_t i = ((size_t)blockIdx.x * blockDim.x + threadIdx.x) * 8;
  float4 v0 = *reinterpret_cast<const float4*>(x + i);
  float4 v1 = *reinterpret_cast<const float4*>(x + i + 4);
  uint4 q;
  q.x = pack_bf16(v0.x, v0.y);
  q.y = pack_bf16(v0.z, v0.w);
  q.z = pack_bf16(v1.x, v1.y);
  q.w = pack_bf16(v1.z, v1.w);
  *reinterpret_cast<uint4*>(xbf + i) = q;
}

__global__ void mask_convert_A(const float* __restrict__ A_ref, const float* __restrict__ sA,
                               const unsigned* __restrict__ thr,
                               __bf16* __restrict__ refA, float* __restrict__ outA) {
  const unsigned T = thr[0];
  int i = blockIdx.x * blockDim.x + threadIdx.x;  // exactly NA threads
  float a = A_ref[i];
  unsigned bits = __float_as_uint(fabsf(sA[i]));
  refA[i] = (__bf16)((bits >= T) ? a : 0.0f);
  outA[i] = a;
}

__global__ void mask_convert_B(const float* __restrict__ B_ref, const float* __restrict__ sB,
                               const unsigned* __restrict__ thr,
                               __bf16* __restrict__ refBt, float* __restrict__ outB) {
  const unsigned T = thr[0];
  int i = blockIdx.x * blockDim.x + threadIdx.x;  // exactly NB threads
  int c = i >> 9;
  int d = i & 511;
  float bv = B_ref[i];
  unsigned bits = __float_as_uint(fabsf(sB[i]));
  refBt[(size_t)d * DIM_C + c] = (__bf16)((bits >= T) ? bv : 0.0f);
  outB[i] = bv;
}

// ============ GEMM1: temp[b,w,c] = sum_h refA[w,h] * xbf[b,c,h] ============
// block tile 128(M=w) x 128(N=c), K-step 32, 8 waves (each 64x32 -> 4x2 frags),
// double-buffered LDS; async global->LDS when the toolchain exposes it.

__global__ __launch_bounds__(256) void gemm1_wmma(const __bf16* __restrict__ refA,
                                                  const __bf16* __restrict__ xbf,
                                                  float* __restrict__ temp) {
  __shared__ __align__(16) __bf16 Ash[2][128 * 32];  // [m][k]
  __shared__ __align__(16) __bf16 Bsh[2][128 * 32];  // [n][k]

  const int n0 = blockIdx.x * 128;   // c tile base
  const int m0 = blockIdx.y * 128;   // w tile base
  const int b  = blockIdx.z;
  const int t  = threadIdx.x;
  const int wv   = t >> 5;
  const int lane = t & 31;
  const int lp   = lane & 15;
  const int half = lane >> 4;
  const int waveM = wv & 1;
  const int waveN = wv >> 1;

  const v8f vzero = {0.f, 0.f, 0.f, 0.f, 0.f, 0.f, 0.f, 0.f};
  v8f acc[4][2];
#pragma unroll
  for (int mt = 0; mt < 4; ++mt)
#pragma unroll
    for (int nt = 0; nt < 2; ++nt) acc[mt][nt] = vzero;

  // staging coordinates: this thread owns 2 contiguous 16B chunks of A and of B
  const int srow = t >> 1;             // tile row 0..127
  const int scq  = (t & 1) * 2;        // chunk pair base (0 or 2) in 8-elem units
  const __bf16* gA = refA + (size_t)(m0 + srow) * DIM_H + scq * 8;
  const __bf16* gB = xbf + (size_t)b * DIM_C * DIM_H + (size_t)(n0 + srow) * DIM_H + scq * 8;
  const int lOff = srow * 32 + scq * 8;

  int cur = 0;

#if USE_ASYNC_COPY
  // prologue: async-fill buffer 0 (ASYNCcnt += 4 per thread)
  __builtin_amdgcn_global_load_async_to_lds_b128(to_global(gA),     to_lds(&Ash[0][lOff]),     0, 0);
  __builtin_amdgcn_global_load_async_to_lds_b128(to_global(gA + 8), to_lds(&Ash[0][lOff + 8]), 0, 0);
  __builtin_amdgcn_global_load_async_to_lds_b128(to_global(gB),     to_lds(&Bsh[0][lOff]),     0, 0);
  __builtin_amdgcn_global_load_async_to_lds_b128(to_global(gB + 8), to_lds(&Bsh[0][lOff + 8]), 0, 0);
#else
  {
    uint4 a0 = *reinterpret_cast<const uint4*>(gA);
    uint4 a1 = *reinterpret_cast<const uint4*>(gA + 8);
    uint4 b0 = *reinterpret_cast<const uint4*>(gB);
    uint4 b1 = *reinterpret_cast<const uint4*>(gB + 8);
    *reinterpret_cast<uint4*>(&Ash[0][lOff])     = a0;
    *reinterpret_cast<uint4*>(&Ash[0][lOff + 8]) = a1;
    *reinterpret_cast<uint4*>(&Bsh[0][lOff])     = b0;
    *reinterpret_cast<uint4*>(&Bsh[0][lOff + 8]) = b1;
  }
  __syncthreads();
#endif

  for (int kk = 0; kk < DIM_H; kk += 32) {
    const bool more = (kk + 32) < DIM_H;

#if USE_ASYNC_COPY
    if (more) {
      const int nb = cur ^ 1;
      __builtin_amdgcn_global_load_async_to_lds_b128(to_global(gA + kk + 32),     to_lds(&Ash[nb][lOff]),     0, 0);
      __builtin_amdgcn_global_load_async_to_lds_b128(to_global(gA + kk + 32 + 8), to_lds(&Ash[nb][lOff + 8]), 0, 0);
      __builtin_amdgcn_global_load_async_to_lds_b128(to_global(gB + kk + 32),     to_lds(&Bsh[nb][lOff]),     0, 0);
      __builtin_amdgcn_global_load_async_to_lds_b128(to_global(gB + kk + 32 + 8), to_lds(&Bsh[nb][lOff + 8]), 0, 0);
      __builtin_amdgcn_s_wait_asynccnt(4);   // current buffer's 4 copies done
    } else {
      __builtin_amdgcn_s_wait_asynccnt(0);
    }
    __syncthreads();                          // everyone's copies visible
#else
    uint4 na0, na1, nb0, nb1;
    if (more) {
      na0 = *reinterpret_cast<const uint4*>(gA + kk + 32);
      na1 = *reinterpret_cast<const uint4*>(gA + kk + 32 + 8);
      nb0 = *reinterpret_cast<const uint4*>(gB + kk + 32);
      nb1 = *reinterpret_cast<const uint4*>(gB + kk + 32 + 8);
    }
#endif

    // ---- compute current buffer ----
    Frag16 a[4], bb[2];
#pragma unroll
    for (int mt = 0; mt < 4; ++mt) {
      int row = waveM * 64 + mt * 16 + lp;
      a[mt].q[0] = *reinterpret_cast<const uint4*>(&Ash[cur][row * 32 + half * 8]);
      a[mt].q[1] = *reinterpret_cast<const uint4*>(&Ash[cur][row * 32 + 16 + half * 8]);
    }
#pragma unroll
    for (int nt = 0; nt < 2; ++nt) {
      int col = waveN * 32 + nt * 16 + lp;
      int base = col * 32 + half * 8;
      bb[nt].q[0] = *reinterpret_cast<const uint4*>(&Bsh[cur][base]);
      bb[nt].q[1] = *reinterpret_cast<const uint4*>(&Bsh[cur][base + 16]);
    }
#pragma unroll
    for (int mt = 0; mt < 4; ++mt)
#pragma unroll
      for (int nt = 0; nt < 2; ++nt)
        acc[mt][nt] = __builtin_amdgcn_wmma_f32_16x16x32_bf16(
            false, a[mt].v, false, bb[nt].v, (short)0, acc[mt][nt], false, false);

#if !USE_ASYNC_COPY
    if (more) {
      const int nb = cur ^ 1;
      *reinterpret_cast<uint4*>(&Ash[nb][lOff])     = na0;
      *reinterpret_cast<uint4*>(&Ash[nb][lOff + 8]) = na1;
      *reinterpret_cast<uint4*>(&Bsh[nb][lOff])     = nb0;
      *reinterpret_cast<uint4*>(&Bsh[nb][lOff + 8]) = nb1;
    }
#endif
    __syncthreads();
    cur ^= 1;
  }

  float* tb = temp + ((size_t)b * DIM_AW + m0) * DIM_C + n0;
#pragma unroll
  for (int mt = 0; mt < 4; ++mt)
#pragma unroll
    for (int nt = 0; nt < 2; ++nt)
#pragma unroll
      for (int r = 0; r < 8; ++r) {
        int gm = waveM * 64 + mt * 16 + half * 8 + r;
        int gn = waveN * 32 + nt * 16 + lp;
        tb[(size_t)gm * DIM_C + gn] = acc[mt][nt][r];
      }
}

// ============ GEMM2: out[(b,w),d] = sum_c temp[(b,w),c] * refBt[d,c] ============

__global__ __launch_bounds__(256) void gemm2_wmma(const float* __restrict__ temp,
                                                  const __bf16* __restrict__ refBt,
                                                  float* __restrict__ out) {
  __shared__ __align__(16) __bf16 Ash[128 * 32];  // [m][k]
  __shared__ __align__(16) __bf16 Bsh[128 * 32];  // [n=d][k=c]

  const int n0 = blockIdx.x * 128;
  const int m0 = blockIdx.y * 128;
  const int t  = threadIdx.x;
  const int wv   = t >> 5;
  const int lane = t & 31;
  const int lp   = lane & 15;
  const int half = lane >> 4;
  const int waveM = wv & 1;
  const int waveN = wv >> 1;

  const v8f vzero = {0.f, 0.f, 0.f, 0.f, 0.f, 0.f, 0.f, 0.f};
  v8f acc[4][2];
#pragma unroll
  for (int mt = 0; mt < 4; ++mt)
#pragma unroll
    for (int nt = 0; nt < 2; ++nt) acc[mt][nt] = vzero;

  for (int kk = 0; kk < DIM_C; kk += 32) {
    // A tile from temp (f32 -> bf16 inline)
#pragma unroll
    for (int i = 0; i < 4; ++i) {
      int idx = t * 4 + i;
      int m  = idx >> 3;
      int kq = idx & 7;
      float4 v = *reinterpret_cast<const float4*>(temp + (size_t)(m0 + m) * DIM_C + kk + kq * 4);
      unsigned* dst = reinterpret_cast<unsigned*>(&Ash[m * 32 + kq * 4]);
      dst[0] = pack_bf16(v.x, v.y);
      dst[1] = pack_bf16(v.z, v.w);
    }
    // B tile from pre-transposed bf16 refBt (contiguous K)
#pragma unroll
    for (int i = 0; i < 2; ++i) {
      int idx = t * 2 + i;
      int r  = idx >> 2;
      int cq = idx & 3;
      uint4 v = *reinterpret_cast<const uint4*>(refBt + (size_t)(n0 + r) * DIM_C + kk + cq * 8);
      *reinterpret_cast<uint4*>(&Bsh[r * 32 + cq * 8]) = v;
    }
    __syncthreads();

    Frag16 a[4], bb[2];
#pragma unroll
    for (int mt = 0; mt < 4; ++mt) {
      int row = waveM * 64 + mt * 16 + lp;
      a[mt].q[0] = *reinterpret_cast<const uint4*>(&Ash[row * 32 + half * 8]);
      a[mt].q[1] = *reinterpret_cast<const uint4*>(&Ash[row * 32 + 16 + half * 8]);
    }
#pragma unroll
    for (int nt = 0; nt < 2; ++nt) {
      int col = waveN * 32 + nt * 16 + lp;
      int base = col * 32 + half * 8;
      bb[nt].q[0] = *reinterpret_cast<const uint4*>(&Bsh[base]);
      bb[nt].q[1] = *reinterpret_cast<const uint4*>(&Bsh[base + 16]);
    }
#pragma unroll
    for (int mt = 0; mt < 4; ++mt)
#pragma unroll
      for (int nt = 0; nt < 2; ++nt)
        acc[mt][nt] = __builtin_amdgcn_wmma_f32_16x16x32_bf16(
            false, a[mt].v, false, bb[nt].v, (short)0, acc[mt][nt], false, false);
    __syncthreads();
  }

  float* ob = out + (size_t)m0 * DIM_BH + n0;
#pragma unroll
  for (int mt = 0; mt < 4; ++mt)
#pragma unroll
    for (int nt = 0; nt < 2; ++nt)
#pragma unroll
      for (int r = 0; r < 8; ++r) {
        int gm = waveM * 64 + mt * 16 + half * 8 + r;
        int gn = waveN * 32 + nt * 16 + lp;
        ob[(size_t)gm * DIM_BH + gn] = acc[mt][nt][r];
      }
}

// ============ launch ============

extern "C" void kernel_launch(void* const* d_in, const int* in_sizes, int n_in,
                              void* d_out, int out_size, void* d_ws, size_t ws_size,
                              hipStream_t stream) {
  const float* x     = (const float*)d_in[0];
  const float* A_ref = (const float*)d_in[1];
  const float* B_ref = (const float*)d_in[2];
  const float* sA    = (const float*)d_in[3];
  const float* sB    = (const float*)d_in[4];
  float* out = (float*)d_out;
  char*  ws  = (char*)d_ws;

  __bf16*   refA  = (__bf16*)(ws + WS_REFA);
  __bf16*   xbf   = (__bf16*)(ws + WS_XBF);
  __bf16*   refBt = (__bf16*)(ws + WS_REFBT);
  float*    temp  = (float*)(ws + WS_TEMP);
  unsigned* hAhi  = (unsigned*)(ws + WS_HIST_AHI);
  unsigned* hAlo  = (unsigned*)(ws + WS_HIST_ALO);
  unsigned* hBhi  = (unsigned*)(ws + WS_HIST_BHI);
  unsigned* hBlo  = (unsigned*)(ws + WS_HIST_BLO);
  unsigned* scal  = (unsigned*)(ws + WS_SCAL);

  (void)hipMemsetAsync(ws + WS_HIST_AHI, 0, WS_ZERO_BYTES, stream);

  // exact top-half thresholds on |scores| via 2-level radix select
  hist_hi<<<2048, 256, 0, stream>>>(sA, NA, hAhi);
  select_scan<<<1, 64, 0, stream>>>(hAhi, JA, (const unsigned*)nullptr, scal + 0);
  hist_lo<<<2048, 256, 0, stream>>>(sA, NA, scal + 0, hAlo);
  select_scan<<<1, 64, 0, stream>>>(hAlo, JA, scal + 0, scal + 2);

  hist_hi<<<256, 256, 0, stream>>>(sB, NB, hBhi);
  select_scan<<<1, 64, 0, stream>>>(hBhi, JB, (const unsigned*)nullptr, scal + 3);
  hist_lo<<<256, 256, 0, stream>>>(sB, NB, scal + 3, hBlo);
  select_scan<<<1, 64, 0, stream>>>(hBlo, JB, scal + 3, scal + 5);

  // one-shot conversions (+ passthrough tuple copies)
  convert_x_bf16<<<NX / (8 * 256), 256, 0, stream>>>(x, xbf);
  mask_convert_A<<<NA / 256, 256, 0, stream>>>(A_ref, sA, scal + 2, refA, out + OUT_ACOPY);
  mask_convert_B<<<NB / 256, 256, 0, stream>>>(B_ref, sB, scal + 5, refBt, out + OUT_BCOPY);

  // GEMM1: [32] x (1024x4096) . (4096x256) -> temp (double-buffered, async LDS fill if available)
  gemm1_wmma<<<dim3(DIM_C / 128, DIM_AW / 128, DIM_B), 256, 0, stream>>>(refA, xbf, temp);
  // GEMM2: (32768x256) . (256x512) -> out
  gemm2_wmma<<<dim3(DIM_BH / 128, (DIM_B * DIM_AW) / 128), 256, 0, stream>>>(temp, refBt, out);
}
